// DOAModel_57191784513920
// MI455X (gfx1250) — matrix operand order repaired and checked by reading, not compiled, not run on hardware
//
#include <hip/hip_runtime.h>
#include <hip/hip_bf16.h>
#include <cstdint>
#include <cstddef>

// ---------------------------------------------------------------------------
// CDNA5 (gfx1250) DOAModel forward.  wave32, WMMA bf16 16x16x32, f32 accum.
// ---------------------------------------------------------------------------

typedef __bf16 bf16_t;
typedef __attribute__((ext_vector_type(16))) __bf16 v16bf;
typedef __attribute__((ext_vector_type(8)))  float  v8f;

static_assert(sizeof(bf16_t) == 2, "bf16 must be 2 bytes");

#define BN_RSQRT 0.9999950000374997f /* 1/sqrt(1+1e-5) */

// ---------------------------------------------------------------------------
// fp32 -> bf16 cast
// ---------------------------------------------------------------------------
__global__ __launch_bounds__(256) void k_cast_f32_bf16(
    const float* __restrict__ in, bf16_t* __restrict__ out, int n) {
  int i = blockIdx.x * blockDim.x + threadIdx.x;
  if (i < n) out[i] = (bf16_t)in[i];
}

// ---------------------------------------------------------------------------
// Implicit-GEMM conv inner step.  KS is compile-time (1 or 3) so the K ->
// (cin,kh,kw) decomposition lowers to mulhi+shift, not runtime division.
// TAIL=false drops all K-range guards (used when kb+32 <= K).
// GUARD=false is the interior fast path: no image-boundary checks, loads from
// a per-lane base pointer so e-offsets fold into the VMEM instruction offset.
// WMMA layouts (ISA 7.12.2, wave32):
//   A 16x32 bf16 : lane l, elem e -> m = l&15, k = (e>>3)*16 + (l>>4)*8 + (e&7)
//   B 32x16 bf16 : lane l, elem e -> n = l&15, k = (e>>3)*16 + (l>>4)*8 + (e&7)
//   D 16x16 f32  : lane l, vgpr v -> n = l&15, m = v + (l>>4)*8
// ---------------------------------------------------------------------------
template <int KS, bool TAIL, bool GUARD>
__device__ __forceinline__ v8f conv_mma_step(
    const bf16_t* __restrict__ xin,   // image base  (guarded path)
    const bf16_t* __restrict__ xfast, // per-lane patch base (fast path)
    const float* __restrict__ wrow,
    int kb, int K, bool ocv,
    int lh, int oh, int ow, int H, int W, int HW, int pad, v8f acc) {
  constexpr int KK = KS * KS;
  v16bf a, b;
#pragma unroll
  for (int e = 0; e < 16; ++e) {
    int ka = kb + ((e >> 3) << 4) + (lh << 3) + (e & 7);
    bool ok = ocv && (!TAIL || ka < K);
    a[e] = ok ? (bf16_t)wrow[ka] : (bf16_t)0.f;
  }
#pragma unroll
  for (int e = 0; e < 16; ++e) {
    int kbi = kb + ((e >> 3) << 4) + (lh << 3) + (e & 7);
    int cin = kbi / KK;          // KK is constexpr -> mulhi+shift
    int rr  = kbi - cin * KK;
    int kh  = rr / KS;
    int kw  = rr - kh * KS;
    bf16_t v = (bf16_t)0.f;
    if (!GUARD) {
      if (!TAIL || kbi < K)
        v = xfast[(size_t)cin * HW + (size_t)kh * W + kw];
    } else {
      if (!TAIL || kbi < K) {
        int ih = oh + kh - pad;
        int iw = ow + kw - pad;
        if (KS == 1 || (ih >= 0 && ih < H && iw >= 0 && iw < W))
          v = xin[(size_t)cin * HW + (size_t)ih * W + iw];
      }
    }
    b[e] = v;
  }
  return __builtin_amdgcn_wmma_f32_16x16x32_bf16(false, a, false, b,
                                                 (short)0, acc, false, false);
}

// One wave computes a 16(outch) x 16(pixel) tile; K = Cin*KS*KS in 32-chunks.
// A tile is treated as one image row (needs W % 16 == 0; the W=40 convs do not meet it),
// and the boundary condition is wave-uniform -> cheap scalar branch.
// Epilogue fuses bias (+ optional BN scale/shift) (+ optional ReLU).
template <typename OutT, int KS>
__global__ __launch_bounds__(256) void k_conv_wmma(
    const bf16_t* __restrict__ x, const float* __restrict__ w,
    const float* __restrict__ bias,
    const float* __restrict__ g,    // BN gamma or null
    const float* __restrict__ beta, // BN beta  or null
    OutT* __restrict__ out,
    int N, int Cin, int H, int W, int Cout, int pad, int relu,
    int tC, int tP) {
  int wid  = (blockIdx.x * blockDim.x + threadIdx.x) >> 5;
  int lane = threadIdx.x & 31;
  int total = N * tC * tP;
  if (wid >= total) return;

  int n  = wid / (tC * tP);
  int r  = wid % (tC * tP);
  int ocBase = (r / tP) * 16;
  int pBase  = (r % tP) * 16;

  int HW = H * W;
  int K  = Cin * KS * KS;
  int Kfull = K & ~31;
  int lm = lane & 15;
  int lh = lane >> 4;

  int pix = pBase + lm;
  int oh = pBase / W;                 // uniform: row of the tile's first pixel
  int owBase = pBase - oh * W;
  int ow = owBase + lm;
  bool ocv = (ocBase + lm) < Cout;
  const float* wrow = w + (size_t)(ocBase + lm) * K;
  const bf16_t* xin = x + (size_t)n * Cin * HW;
  const bf16_t* xfast = xin + (size_t)(oh - pad) * W + (ow - pad);

  bool interior = (KS == 1) ||
                  (oh >= pad && oh + pad < H &&
                   owBase >= pad && owBase + 15 + pad < W);

  v8f acc = {};
  int kb = 0;
  if (interior) {
    for (; kb < Kfull; kb += 32) {
      if (kb + 32 < K)
        __builtin_prefetch(&wrow[kb + 32], 0, 0); // global_prefetch_b8
      acc = conv_mma_step<KS, false, false>(xin, xfast, wrow, kb, K, ocv, lh,
                                            oh, ow, H, W, HW, pad, acc);
    }
    if (kb < K)
      acc = conv_mma_step<KS, true, false>(xin, xfast, wrow, kb, K, ocv, lh,
                                           oh, ow, H, W, HW, pad, acc);
  } else {
    for (; kb < Kfull; kb += 32) {
      if (kb + 32 < K)
        __builtin_prefetch(&wrow[kb + 32], 0, 0);
      acc = conv_mma_step<KS, false, true>(xin, xfast, wrow, kb, K, ocv, lh,
                                           oh, ow, H, W, HW, pad, acc);
    }
    if (kb < K)
      acc = conv_mma_step<KS, true, true>(xin, xfast, wrow, kb, K, ocv, lh,
                                          oh, ow, H, W, HW, pad, acc);
  }

  // ---- epilogue: bias + BN + ReLU ----
#pragma unroll
  for (int v = 0; v < 8; ++v) {
    int oc = ocBase + v + (lh << 3);
    if (oc < Cout) {
      float val = acc[v] + bias[oc];
      if (g) val = val * (g[oc] * BN_RSQRT) + beta[oc];
      if (relu) val = fmaxf(val, 0.f);
      out[((size_t)n * Cout + oc) * HW + pix] = (OutT)val;
    }
  }
}

// ---------------------------------------------------------------------------
// Correlation GEMM: aff[p][q] = sum_c fn[c,p]*fn[c,q], fn (B,256,1600) bf16.
// Geometry is constexpr so the e*P element offsets fold into the global_load
// 24-bit instruction offsets.  Epilogue fuses zero-window mask and alpha:
//   s = aff * (1 - exp(-(dh^2+dw^2)/8)) * alpha
// ---------------------------------------------------------------------------
__global__ __launch_bounds__(256) void k_corr_gemm(
    const bf16_t* __restrict__ fn, float* __restrict__ s,
    const float* __restrict__ alphap, int B) {
  constexpr int C = 256, P = 1600, Wd = 40, TP = P / 16;
  int wid  = (blockIdx.x * blockDim.x + threadIdx.x) >> 5;
  int lane = threadIdx.x & 31;
  int total = B * TP * TP;
  if (wid >= total) return;

  int b  = wid / (TP * TP);
  int r  = wid % (TP * TP);
  int pB = (r / TP) * 16;
  int qB = (r % TP) * 16;
  int lm = lane & 15, lh = lane >> 4;

  const bf16_t* f = fn + (size_t)b * C * P;
  const bf16_t* fa = f + (size_t)(lh * 16) * P + pB + lm; // A: k-base lh*16
  const bf16_t* fb = f + (size_t)(lh * 16) * P + qB + lm; // B: k-base lh*16
  v8f acc = {};

  for (int cb = 0; cb < C; cb += 32) {
    const bf16_t* pa = fa + (size_t)cb * P;
    const bf16_t* pb = fb + (size_t)cb * P;
    v16bf a, bm;
#pragma unroll
    for (int e = 0; e < 16; ++e) {
      // k = cb + lh*16 + e
      a[e] = pa[(size_t)e * P];
    }
#pragma unroll
    for (int e = 0; e < 16; ++e) {
      // k = cb + lh*16 + e
      bm[e] = pb[(size_t)e * P];
    }
    acc = __builtin_amdgcn_wmma_f32_16x16x32_bf16(false, a, false, bm,
                                                  (short)0, acc, false, false);
  }

  float alpha = *alphap;
  int q  = qB + lm;
  int qh = q / Wd, qw = q - (q / Wd) * Wd;
#pragma unroll
  for (int v = 0; v < 8; ++v) {
    int p  = pB + v + (lh << 3);
    int ph = p / Wd, pw = p - (p / Wd) * Wd;
    float dh = (float)(qh - ph), dw = (float)(qw - pw);
    float mask = 1.f - __expf(-(dh * dh + dw * dw) * 0.125f);
    s[((size_t)b * P + p) * P + q] = acc[v] * mask * alpha;
  }
}

// ---------------------------------------------------------------------------
// softmax stats: one wave per row / column of s (P = 1600)
// ---------------------------------------------------------------------------
__global__ __launch_bounds__(256) void k_row_stats(
    const float* __restrict__ s, float* __restrict__ rmax,
    float* __restrict__ rsum, int B) {
  constexpr int P = 1600;
  int wid  = (blockIdx.x * blockDim.x + threadIdx.x) >> 5;
  int lane = threadIdx.x & 31;
  if (wid >= B * P) return;
  const float* row = s + (size_t)wid * P;
  float m = -3.4e38f;
  for (int q = lane; q < P; q += 32) m = fmaxf(m, row[q]);
#pragma unroll
  for (int o = 16; o; o >>= 1) m = fmaxf(m, __shfl_xor(m, o, 32));
  float sm = 0.f;
  for (int q = lane; q < P; q += 32) sm += __expf(row[q] - m);
#pragma unroll
  for (int o = 16; o; o >>= 1) sm += __shfl_xor(sm, o, 32);
  if (lane == 0) { rmax[wid] = m; rsum[wid] = sm; }
}

__global__ __launch_bounds__(256) void k_col_stats(
    const float* __restrict__ s, float* __restrict__ cmax,
    float* __restrict__ csum, int B) {
  constexpr int P = 1600;
  int wid  = (blockIdx.x * blockDim.x + threadIdx.x) >> 5;
  int lane = threadIdx.x & 31;
  if (wid >= B * P) return;
  int b = wid / P, q = wid % P;
  const float* base = s + (size_t)b * P * P + q;
  float m = -3.4e38f;
  for (int p = lane; p < P; p += 32) m = fmaxf(m, base[(size_t)p * P]);
#pragma unroll
  for (int o = 16; o; o >>= 1) m = fmaxf(m, __shfl_xor(m, o, 32));
  float sm = 0.f;
  for (int p = lane; p < P; p += 32) sm += __expf(base[(size_t)p * P] - m);
#pragma unroll
  for (int o = 16; o; o >>= 1) sm += __shfl_xor(sm, o, 32);
  if (lane == 0) { cmax[wid] = m; csum[wid] = sm; }
}

// ---------------------------------------------------------------------------
// top-20 over p of xc[p,q] = softmax_row * softmax_col, one thread per (b,q).
// Output descending (matches lax.top_k), stored bf16 as (B,20,P).
// ---------------------------------------------------------------------------
__global__ __launch_bounds__(256) void k_topk(
    const float* __restrict__ s,
    const float* __restrict__ rmax, const float* __restrict__ rsum,
    const float* __restrict__ cmax, const float* __restrict__ csum,
    bf16_t* __restrict__ valq, int B) {
  constexpr int P = 1600, K = 20;
  int i = blockIdx.x * blockDim.x + threadIdx.x;
  if (i >= B * P) return;
  int b = i / P, q = i % P;
  const float* base = s + (size_t)b * P * P + q;
  const float* rm = rmax + (size_t)b * P;
  const float* rs = rsum + (size_t)b * P;
  float cm = cmax[i], cs = csum[i];

  float best[K];
#pragma unroll
  for (int k = 0; k < K; ++k) best[k] = -3.4e38f;

  for (int p = 0; p < P; ++p) {
    float sv = base[(size_t)p * P];
    float xc = (__expf(sv - rm[p]) / rs[p]) * (__expf(sv - cm) / cs);
    if (xc > best[K - 1]) {
      int j = K - 1;
      while (j > 0 && best[j - 1] < xc) { best[j] = best[j - 1]; --j; }
      best[j] = xc;
    }
  }
  for (int k = 0; k < K; ++k)
    valq[((size_t)b * K + k) * P + q] = (bf16_t)best[k];
}

// ---------------------------------------------------------------------------
// kxk max-pool (stride k), bf16
// ---------------------------------------------------------------------------
__global__ __launch_bounds__(256) void k_maxpool(
    const bf16_t* __restrict__ in, bf16_t* __restrict__ out,
    int N, int C, int H, int W, int k) {
  int Ho = H / k, Wo = W / k;
  size_t total = (size_t)N * C * Ho * Wo;
  size_t i = (size_t)blockIdx.x * blockDim.x + threadIdx.x;
  if (i >= total) return;
  int ow = i % Wo; size_t t = i / Wo;
  int oh = t % Ho; t /= Ho;
  int c = t % C; int n = t / C;
  const bf16_t* p = in + ((size_t)n * C + c) * H * W;
  float m = -3.4e38f;
  for (int a = 0; a < k; ++a)
    for (int b = 0; b < k; ++b)
      m = fmaxf(m, (float)p[(size_t)(oh * k + a) * W + ow * k + b]);
  out[i] = (bf16_t)m;
}

// ---------------------------------------------------------------------------
// bilinear resize, align_corners=True; writes into channel window [coff, coff+C)
// of an output tensor with Ctot channels (fused concat).
// ---------------------------------------------------------------------------
template <typename TI, typename TO>
__global__ __launch_bounds__(256) void k_bilinear(
    const TI* __restrict__ in, TO* __restrict__ out,
    int N, int C, int Hin, int Win, int Hout, int Wout, int coff, int Ctot) {
  size_t total = (size_t)N * C * Hout * Wout;
  size_t i = (size_t)blockIdx.x * blockDim.x + threadIdx.x;
  if (i >= total) return;
  int ox = i % Wout; size_t t = i / Wout;
  int oy = t % Hout; t /= Hout;
  int c = t % C; int n = t / C;

  float fy = (Hout > 1) ? oy * (float)(Hin - 1) / (float)(Hout - 1) : 0.f;
  float fx = (Wout > 1) ? ox * (float)(Win - 1) / (float)(Wout - 1) : 0.f;
  int y0 = (int)floorf(fy); int y1 = min(y0 + 1, Hin - 1); float wy = fy - y0;
  int x0 = (int)floorf(fx); int x1 = min(x0 + 1, Win - 1); float wx = fx - x0;

  const TI* p = in + ((size_t)n * C + c) * Hin * Win;
  float v00 = (float)p[(size_t)y0 * Win + x0];
  float v01 = (float)p[(size_t)y0 * Win + x1];
  float v10 = (float)p[(size_t)y1 * Win + x0];
  float v11 = (float)p[(size_t)y1 * Win + x1];
  float v = (v00 * (1.f - wx) + v01 * wx) * (1.f - wy) +
            (v10 * (1.f - wx) + v11 * wx) * wy;
  out[((size_t)n * Ctot + coff + c) * Hout * Wout + (size_t)oy * Wout + ox] = (TO)v;
}

// ---------------------------------------------------------------------------
// SPP [4,2,1]: out (N,768,40,40) from q (N,256,40,40); blockwise max + nearest up
// ---------------------------------------------------------------------------
__global__ __launch_bounds__(256) void k_spp(
    const bf16_t* __restrict__ q, bf16_t* __restrict__ out, int N) {
  const int C = 256, H = 40, W = 40;
  size_t total = (size_t)N * 768 * H * W;
  size_t i = (size_t)blockIdx.x * blockDim.x + threadIdx.x;
  if (i >= total) return;
  int ow = i % W; size_t t = i / W;
  int oh = t % H; t /= H;
  int cc = t % 768; int n = t / 768;
  int c, ks;
  if (cc < 256)      { c = cc;       ks = 4; }
  else if (cc < 512) { c = cc - 256; ks = 2; }
  else               { c = cc - 512; ks = 1; }
  int h0 = (oh / ks) * ks, w0 = (ow / ks) * ks;
  const bf16_t* p = q + ((size_t)n * C + c) * H * W;
  float m = -3.4e38f;
  for (int a = 0; a < ks; ++a)
    for (int b = 0; b < ks; ++b)
      m = fmaxf(m, (float)p[(size_t)(h0 + a) * W + w0 + b]);
  out[i] = (bf16_t)m;
}

// ---------------------------------------------------------------------------
// channel-wise L2 normalize: one wave per (b,pixel) over C=256 channels
// ---------------------------------------------------------------------------
__global__ __launch_bounds__(256) void k_normalize(
    const bf16_t* __restrict__ qs, bf16_t* __restrict__ fn, int B) {
  constexpr int C = 256, P = 1600;
  int wid  = (blockIdx.x * blockDim.x + threadIdx.x) >> 5;
  int lane = threadIdx.x & 31;
  if (wid >= B * P) return;
  int b = wid / P, p = wid % P;
  const bf16_t* base = qs + (size_t)b * C * P + p;
  float ss = 0.f;
#pragma unroll
  for (int c = 0; c < C / 32; ++c) {
    float v = (float)base[(size_t)(c * 32 + lane) * P];
    ss += v * v;
  }
#pragma unroll
  for (int o = 16; o; o >>= 1) ss += __shfl_xor(ss, o, 32);
  float inv = 1.f / fmaxf(sqrtf(ss), 1e-12f);
  bf16_t* o = fn + (size_t)b * C * P + p;
#pragma unroll
  for (int c = 0; c < C / 32; ++c)
    o[(size_t)(c * 32 + lane) * P] =
        (bf16_t)((float)base[(size_t)(c * 32 + lane) * P] * inv);
}

// ===========================================================================
// host launcher
// ===========================================================================
extern "C" void kernel_launch(void* const* d_in, const int* in_sizes, int n_in,
                              void* d_out, int out_size, void* d_ws, size_t ws_size,
                              hipStream_t stream) {
  (void)n_in; (void)out_size; (void)ws_size;
  auto F = [&](int i) { return (const float*)d_in[i]; };

  const float* xq = nullptr;
  const float* vggw[12]; const float* vggb[12];
  const float *icw, *icb, *ibg, *ibb;      // in_conv, in_bn
  const float *pcw, *pcb, *pbg, *pbb;      // pyr_conv, pyr_bn
  const float *v1w, *v1b, *vbg, *vbb;      // v1, v_bn
  const float *v2w, *v2b, *v3w, *v3b, *alpha;

  if (in_sizes[0] == 1228800 && in_sizes[2] == 1728) {
    // insertion order: xq, xp, vgg[w,b]*12, in_conv, in_bn, pyr_conv, pyr_bn,
    //                  v1, v_bn, v2, v3, alpha
    xq = F(0);
    int i = 2;
    for (int j = 0; j < 12; ++j) { vggw[j] = F(i++); vggb[j] = F(i++); }
    icw = F(i++); icb = F(i++); ibg = F(i++); ibb = F(i++);
    pcw = F(i++); pcb = F(i++); pbg = F(i++); pbb = F(i++);
    v1w = F(i++); v1b = F(i++); vbg = F(i++); vbb = F(i++);
    v2w = F(i++); v2b = F(i++); v3w = F(i++); v3b = F(i++);
    alpha = F(i++);
  } else if (in_sizes[0] == 1228800) {
    // xq, xp, then jax-sorted params
    xq = F(0);
    alpha = F(2); ibb = F(3); ibg = F(4); icb = F(5); icw = F(6);
    pbb = F(7); pbg = F(8); pcb = F(9); pcw = F(10);
    v1b = F(11); v1w = F(12); v2b = F(13); v2w = F(14);
    v3b = F(15); v3w = F(16); vbb = F(17); vbg = F(18);
    for (int j = 0; j < 12; ++j) { vggb[j] = F(19 + 2 * j); vggw[j] = F(20 + 2 * j); }
  } else {
    // fully jax-sorted: params..., xp, xq
    alpha = F(0); ibb = F(1); ibg = F(2); icb = F(3); icw = F(4);
    pbb = F(5); pbg = F(6); pcb = F(7); pcw = F(8);
    v1b = F(9); v1w = F(10); v2b = F(11); v2w = F(12);
    v3b = F(13); v3w = F(14); vbb = F(15); vbg = F(16);
    for (int j = 0; j < 12; ++j) { vggb[j] = F(17 + 2 * j); vggw[j] = F(18 + 2 * j); }
    xq = F(42);
  }

  // ---- workspace layout (bytes) ----
  char* ws = (char*)d_ws;
  const size_t MB = 1ull << 20;
  bf16_t* PING = (bf16_t*)(ws + 0);        // 56 MB (max act: 4*64*320*320 bf16)
  bf16_t* PONG = (bf16_t*)(ws + 56 * MB);  // 56 MB
  bf16_t* X1   = (bf16_t*)(ws + 112 * MB); // (4,128,80,80)
  bf16_t* X2   = (bf16_t*)(ws + 120 * MB); // (4,256,40,40)
  bf16_t* X3   = (bf16_t*)(ws + 124 * MB); // (4,512,20,20)
  bf16_t* FEAT = (bf16_t*)(ws + 126 * MB); // (4,896,40,40)
  bf16_t* Qb   = (bf16_t*)(ws + 138 * MB); // (4,256,40,40)
  bf16_t* SPPb = (bf16_t*)(ws + 142 * MB); // (4,768,40,40)
  bf16_t* QSb  = (bf16_t*)(ws + 152 * MB); // (4,256,40,40)
  bf16_t* FNb  = (bf16_t*)(ws + 156 * MB); // (4,256,1600)
  // after pyr_conv, VGG ping/pong are dead -> reuse:
  float* Smat = (float*)PING;              // (4,1600,1600) f32 = 41 MB
  float* rmax = (float*)(ws + 56 * MB);
  float* rsum = rmax + 8192;
  float* cmax = rsum + 8192;
  float* csum = cmax + 8192;
  bf16_t* VALQ = (bf16_t*)(ws + 57 * MB);  // (4,20,1600)
  bf16_t* O1   = (bf16_t*)(ws + 58 * MB);  // (4,16,1600)
  bf16_t* O2   = (bf16_t*)(ws + 59 * MB);  // (4,16,1600)
  float*  O3   = (float*)(ws + 60 * MB);   // (4,1,1600)

  const int N = 4;
  auto cdiv = [](long a, long b) { return (int)((a + b - 1) / b); };

  auto conv = [&](const bf16_t* x, const float* w, const float* b,
                  const float* g, const float* be, void* out, bool f32out,
                  int Cin, int H, int W, int Cout, int ks, int pad, bool relu) {
    int HW = H * W;
    int tC = (Cout + 15) / 16, tP = (HW + 15) / 16;
    long tiles = (long)N * tC * tP;
    int blocks = cdiv(tiles, 8); // 8 waves (256 threads) per block
    if (ks == 3) {
      if (f32out)
        k_conv_wmma<float, 3><<<blocks, 256, 0, stream>>>(
            x, w, b, g, be, (float*)out, N, Cin, H, W, Cout, pad, relu, tC, tP);
      else
        k_conv_wmma<bf16_t, 3><<<blocks, 256, 0, stream>>>(
            x, w, b, g, be, (bf16_t*)out, N, Cin, H, W, Cout, pad, relu, tC, tP);
    } else {
      if (f32out)
        k_conv_wmma<float, 1><<<blocks, 256, 0, stream>>>(
            x, w, b, g, be, (float*)out, N, Cin, H, W, Cout, pad, relu, tC, tP);
      else
        k_conv_wmma<bf16_t, 1><<<blocks, 256, 0, stream>>>(
            x, w, b, g, be, (bf16_t*)out, N, Cin, H, W, Cout, pad, relu, tC, tP);
    }
  };
  auto pool = [&](const bf16_t* in, bf16_t* out, int C, int H, int W, int k) {
    long tot = (long)N * C * (H / k) * (W / k);
    k_maxpool<<<cdiv(tot, 256), 256, 0, stream>>>(in, out, N, C, H, W, k);
  };
  auto bil = [&](const bf16_t* in, bf16_t* out, int C, int Hin, int Win,
                 int coff, int Ctot) {
    long tot = (long)N * C * 40 * 40;
    k_bilinear<bf16_t, bf16_t><<<cdiv(tot, 256), 256, 0, stream>>>(
        in, out, N, C, Hin, Win, 40, 40, coff, Ctot);
  };

  // ---- VGG19 features[:28] ----
  { int n = 4 * 3 * 320 * 320;
    k_cast_f32_bf16<<<cdiv(n, 256), 256, 0, stream>>>(xq, PING, n); }
  conv(PING, vggw[0], vggb[0], nullptr, nullptr, PONG, false, 3,   320, 320, 64,  3, 1, true);
  conv(PONG, vggw[1], vggb[1], nullptr, nullptr, PING, false, 64,  320, 320, 64,  3, 1, true);
  pool(PING, PONG, 64, 320, 320, 2);                                   // -> 160
  conv(PONG, vggw[2], vggb[2], nullptr, nullptr, PING, false, 64,  160, 160, 128, 3, 1, true);
  conv(PING, vggw[3], vggb[3], nullptr, nullptr, PONG, false, 128, 160, 160, 128, 3, 1, true);
  pool(PONG, X1, 128, 160, 160, 2);                                    // x1 (128,80,80)
  conv(X1,   vggw[4], vggb[4], nullptr, nullptr, PING, false, 128, 80, 80, 256, 3, 1, true);
  conv(PING, vggw[5], vggb[5], nullptr, nullptr, PONG, false, 256, 80, 80, 256, 3, 1, true);
  conv(PONG, vggw[6], vggb[6], nullptr, nullptr, PING, false, 256, 80, 80, 256, 3, 1, true);
  conv(PING, vggw[7], vggb[7], nullptr, nullptr, PONG, false, 256, 80, 80, 256, 3, 1, true);
  pool(PONG, X2, 256, 80, 80, 2);                                      // x2 (256,40,40)
  conv(X2,   vggw[8],  vggb[8],  nullptr, nullptr, PING, false, 256, 40, 40, 512, 3, 1, true);
  conv(PING, vggw[9],  vggb[9],  nullptr, nullptr, PONG, false, 512, 40, 40, 512, 3, 1, true);
  conv(PONG, vggw[10], vggb[10], nullptr, nullptr, PING, false, 512, 40, 40, 512, 3, 1, true);
  conv(PING, vggw[11], vggb[11], nullptr, nullptr, PONG, false, 512, 40, 40, 512, 3, 1, true);
  pool(PONG, X3, 512, 40, 40, 2);                                      // x3 (512,20,20)

  // ---- multi-scale feature concat (fused into bilinear writes) ----
  bil(X1, FEAT, 128, 80, 80, 0,   896);
  bil(X2, FEAT, 256, 40, 40, 128, 896);
  bil(X3, FEAT, 512, 20, 20, 384, 896);

  // ---- in_conv + BN + ReLU ----
  conv(FEAT, icw, icb, ibg, ibb, Qb, false, 896, 40, 40, 256, 3, 1, true);

  // ---- SPP [4,2,1] ----
  { long tot = (long)N * 768 * 1600;
    k_spp<<<cdiv(tot, 256), 256, 0, stream>>>(Qb, SPPb, N); }

  // ---- pyr_conv + BN + ReLU ----
  conv(SPPb, pcw, pcb, pbg, pbb, QSb, false, 768, 40, 40, 256, 3, 1, true);

  // ---- correlation ----
  { int waves = N * 1600;
    k_normalize<<<cdiv((long)waves * 32, 256), 256, 0, stream>>>(QSb, FNb, N); }
  { long tiles = (long)N * 100 * 100;
    k_corr_gemm<<<cdiv(tiles, 8), 256, 0, stream>>>(FNb, Smat, alpha, N); }
  { int waves = N * 1600; int blocks = cdiv((long)waves * 32, 256);
    k_row_stats<<<blocks, 256, 0, stream>>>(Smat, rmax, rsum, N);
    k_col_stats<<<blocks, 256, 0, stream>>>(Smat, cmax, csum, N); }
  { int th = N * 1600;
    k_topk<<<cdiv(th, 256), 256, 0, stream>>>(Smat, rmax, rsum, cmax, csum,
                                              VALQ, N); }

  // ---- value head ----
  conv(VALQ, v1w, v1b, vbg, vbb, O1, false, 20, 40, 40, 16, 3, 1, true);
  conv(O1,   v2w, v2b, nullptr, nullptr, O2, false, 16, 40, 40, 16, 3, 1, false);
  conv(O2,   v3w, v3b, nullptr, nullptr, O3, true,  16, 40, 40, 1,  1, 0, false);

  // ---- final bilinear to (4,1,320,320) ----
  { long tot = (long)N * 1 * 320 * 320;
    k_bilinear<float, float><<<cdiv(tot, 256), 256, 0, stream>>>(
        O3, (float*)d_out, N, 1, 40, 40, 320, 320, 0, 1); }
}
